// BinReLUConvBN_62362925138592
// MI455X (gfx1250) — compile-verified, matrix operand order checked by reading, not run on hardware
//
#include <hip/hip_runtime.h>
#include <hip/hip_bf16.h>

// ---------------- problem constants ----------------
constexpr int kNB   = 64;          // batch
constexpr int kC    = 256;         // channels in/out
constexpr int kH    = 56;
constexpr int kW    = 56;
constexpr int kHW   = kH * kW;     // 3136
constexpr int kHP   = 58;          // padded H
constexpr int kWP   = 58;          // padded W
constexpr int kKTOT = kC * 9;      // 2304 GEMM-K
constexpr int kMTOT = kNB * kHW;   // 200704 GEMM-M
constexpr float kEPS = 1e-5f;

// B staging: per conv tap, per block: 8 kt-rows x 8 ntg x 32 lanes x 16 elems x 2B
constexpr int kTapBytes = 8 * 8 * 32 * 16 * 2;   // 65536

typedef __attribute__((ext_vector_type(16))) __bf16 bf16x16;
typedef __attribute__((ext_vector_type(8)))  float  f32x8;
typedef unsigned int uintx4 __attribute__((ext_vector_type(4)));
typedef int          intx4  __attribute__((ext_vector_type(4)));
typedef int          intx8  __attribute__((ext_vector_type(8)));

#if defined(__has_builtin)
#  if __has_builtin(__builtin_amdgcn_tensor_load_to_lds)
#    define USE_TDM 1
#  endif
#endif
#ifndef USE_TDM
#  define USE_TDM 0
#endif

union Frag {
    bf16x16 v;
    float4  q[2];   // two 16-byte halves
};

// ---------------- kernel 1: BN statistics -> scale/shift ----------------
__global__ void bn_stats_kernel(const float* __restrict__ x,
                                const float* __restrict__ gamma,
                                const float* __restrict__ beta,
                                float* __restrict__ scale,
                                float* __restrict__ shift) {
    __shared__ float s1[256];
    __shared__ float s2[256];
    const int c = blockIdx.x;
    const int t = threadIdx.x;
    float a = 0.f, b = 0.f;
    for (int n = 0; n < kNB; ++n) {
        const float* p = x + ((size_t)n * kC + c) * kHW;
        for (int i = t; i < kHW; i += 256) {
            float v = p[i];
            a += v;
            b += v * v;
        }
    }
    s1[t] = a; s2[t] = b;
    __syncthreads();
    for (int s = 128; s > 0; s >>= 1) {
        if (t < s) { s1[t] += s1[t + s]; s2[t] += s2[t + s]; }
        __syncthreads();
    }
    if (t == 0) {
        const float cnt  = (float)(kNB * kHW);
        const float mean = s1[0] / cnt;
        const float var  = s2[0] / cnt - mean * mean;
        const float sc   = gamma[c] * rsqrtf(var + kEPS);
        scale[c] = sc;
        shift[c] = beta[c] - mean * sc;
    }
}

// ---------------- kernel 2: binarize -> padded NHWC bf16 ----------------
__global__ void binarize_kernel(const float* __restrict__ x,
                                const float* __restrict__ scale,
                                const float* __restrict__ shift,
                                __bf16* __restrict__ xpad) {
    const int pos = blockIdx.x * blockDim.x + threadIdx.x;   // 0..kMTOT-1
    if (pos >= kMTOT) return;
    const int c0 = blockIdx.y * 8;
    const int n  = pos / kHW, hw = pos % kHW;
    const int h  = hw / kW,   w  = hw % kW;

    __attribute__((aligned(16))) __bf16 vals[8];
#pragma unroll
    for (int j = 0; j < 8; ++j) {
        const int c = c0 + j;
        const float v  = x[((size_t)n * kC + c) * kHW + hw];
        const float xn = scale[c] * v + shift[c];
        vals[j] = (__bf16)(xn >= 0.f ? 1.f : -1.f);
    }
    const size_t dst = (((size_t)n * kHP + (h + 1)) * kWP + (w + 1)) * kC + c0;
    *(float4*)(xpad + dst) = *(const float4*)vals;
}

// ---------------- kernel 3: repack weights into B-fragment layout ----------
// GEMM K ordering: K = (kh*3+kw)*256 + ci ; kt = K/32 (72 tiles), nt = co/16.
// Wb[((kt*16 + nt)*32 + lane)*16 + e] = B[kt*32 + kOf(lane,e)][nt*16 + (lane&15)]
__global__ void wcvt_kernel(const float* __restrict__ Wsrc,
                            __bf16* __restrict__ Wb) {
    const int idx = blockIdx.x * blockDim.x + threadIdx.x;   // kt*16*32 + nt*32 + lane
    if (idx >= 72 * 16 * 32) return;
    const int lane  = idx & 31;
    const int nt    = (idx >> 5) & 15;
    const int kt    = idx >> 9;
    const int khalf = (lane >= 16) ? 8 : 0;
    const int co    = nt * 16 + (lane & 15);

    __attribute__((aligned(32))) __bf16 vals[16];
#pragma unroll
    for (int e = 0; e < 16; ++e) {
        const int kof  = (e < 8) ? (khalf + e) : (khalf + 8 + e); // +16 for hi 8
        const int KK   = kt * 32 + kof;
        const int khkw = KK >> 8;      // tap index 0..8
        const int ci   = KK & 255;
        vals[e] = (__bf16)Wsrc[((size_t)co * kC + ci) * 9 + khkw];
    }
    __bf16* dst = Wb + (size_t)idx * 16;
    *(float4*)(dst)     = *(const float4*)(vals);
    *(float4*)(dst + 8) = *(const float4*)(vals + 8);
}

// ---------------- kernel 4: implicit-GEMM conv via bf16 WMMA -------------
// Block: 256 threads = 8 waves (2 wave-rows x 4 wave-cols).
// Wave tile: 64 M x 32 N (4x2 WMMA tiles). Block tile: 128 M x 128 N.
// B tiles staged per conv tap into LDS by the Tensor Data Mover (double
// buffered, overlapped with WMMA on the previous tap).
__global__ void __launch_bounds__(256, 1)
wmma_conv_kernel(const __bf16* __restrict__ xpad,
                 const __bf16* __restrict__ Wb,
                 float* __restrict__ out) {
    extern __shared__ char smem[];   // [2*kTapBytes B-stage][8KB epilogue]

    const int tid   = threadIdx.x;
    const int lane  = tid & 31;
    const int wave  = tid >> 5;
    const int waveM = wave & 1;
    const int waveN = wave >> 1;
    const int lrow  = lane & 15;
    const int hiA   = lane >> 4;                    // 0 or 1: A/C half select

    const int M0      = blockIdx.x * 128 + waveM * 64;
    const int coBase  = blockIdx.y * 128 + waveN * 32;
    const int ntgBase = blockIdx.y * 8;             // block's first co-group

    // per-lane row bookkeeping for the 4 M tiles
    int posBase[4], nIdx[4], hwIdx[4];
#pragma unroll
    for (int mt = 0; mt < 4; ++mt) {
        const int M = M0 + mt * 16 + lrow;
        const int n = M / kHW, hw = M % kHW;
        const int h = hw / kW, w = hw % kW;
        nIdx[mt] = n; hwIdx[mt] = hw;
        // padded coord for tap (kh,kw) is (h+kh, w+kw): +1 pad and -1 tap cancel
        posBase[mt] = ((n * kHP + h) * kWP + w) * kC;
    }

    // ---- B staging: one tensor_load_to_lds per tap (64KB, 8 rows x 8KB) ----
    auto stageB = [&](int tap, int buf) {
        const size_t goff = ((size_t)(tap * 8 * 16 + ntgBase)) * 512; // elems
#if USE_TDM
        if (wave == 0) {
            const unsigned long long ga =
                (unsigned long long)(const void*)(Wb + goff);
            const unsigned ldsOff =
                (unsigned)(unsigned long long)(smem + buf * kTapBytes);
            uintx4 g0;
            g0.x = 1u;                                   // count=1 (valid D#)
            g0.y = ldsOff;                               // lds_addr
            g0.z = (unsigned)(ga & 0xffffffffull);       // global_addr lo
            g0.w = (unsigned)((ga >> 32) & 0x01ffffffull) | (2u << 30); // type=2
            intx8 g1;
            g1[0] = 0x00010000;          // workgroup_mask=0 | data_size=2B
            g1[1] = (int)(4096u << 16);  // tensor_dim0 = 4096 elems (lo16@b48)
            g1[2] = (int)(8u << 16);     // tensor_dim1 = 8 (lo16@b80)
            g1[3] = (int)(4096u << 16);  // tile_dim0 = 4096 (@b112)
            g1[4] = 8;                   // tile_dim1 = 8, tile_dim2 = 0
            g1[5] = 8192;                // tensor_dim0_stride = 8192 elems
            g1[6] = 0;
            g1[7] = 0;
            intx4 gz = {0, 0, 0, 0};
#if __clang_major__ >= 23
            intx8 gz8 = {0, 0, 0, 0, 0, 0, 0, 0};
            __builtin_amdgcn_tensor_load_to_lds(g0, g1, gz, gz, gz8, 0);
#else
            __builtin_amdgcn_tensor_load_to_lds(g0, g1, gz, gz, 0);
#endif
        }
#else
        // cooperative fallback: 4096 x 16B moves, 16 per thread
        for (int it = 0; it < 16; ++it) {
            const int e16 = it * 256 + tid;        // which 16B chunk of 64KB
            const int ktl = e16 >> 9;              // 512 chunks per kt-row
            const int j   = e16 & 511;
            const float4 v = *(const float4*)((const char*)Wb +
                                (goff + (size_t)ktl * 8192) * 2 + (size_t)j * 16);
            *(float4*)(smem + buf * kTapBytes + (ktl * 512 + j) * 16) = v;
        }
#endif
    };
    auto waitB = [&]() {
#if USE_TDM
        if (wave == 0) __builtin_amdgcn_s_wait_tensorcnt(0);
#endif
        __syncthreads();
    };

    const f32x8 zero = {0.f, 0.f, 0.f, 0.f, 0.f, 0.f, 0.f, 0.f};
    f32x8 acc[4][2];
#pragma unroll
    for (int mt = 0; mt < 4; ++mt)
#pragma unroll
        for (int nt = 0; nt < 2; ++nt) acc[mt][nt] = zero;

    stageB(0, 0);
    waitB();

#pragma unroll 1
    for (int tap = 0; tap < 9; ++tap) {
        const int buf = tap & 1;
        if (tap < 8) stageB(tap + 1, buf ^ 1);   // async TDM overlaps compute

        const int roff = ((tap / 3) * kWP + (tap % 3)) * kC;
        if (tap < 8) {
            const int nroff = (((tap + 1) / 3) * kWP + ((tap + 1) % 3)) * kC;
#pragma unroll
            for (int mt = 0; mt < 4; ++mt)
                __builtin_prefetch(xpad + posBase[mt] + nroff, 0, 0);
        }

        const char* bstage = smem + buf * kTapBytes;
#pragma unroll 1
        for (int kc = 0; kc < 8; ++kc) {
            const int ci0 = kc * 32;
            Frag a[4], b[2];
#pragma unroll
            for (int mt = 0; mt < 4; ++mt) {
                const __bf16* p = xpad + posBase[mt] + roff + ci0 + hiA * 8;
                a[mt].q[0] = *(const float4*)(p);        // K 0..7  (or 8..15)
                a[mt].q[1] = *(const float4*)(p + 16);   // K 16..23 (or 24..31)
            }
#pragma unroll
            for (int nt = 0; nt < 2; ++nt) {
                const int ntgl = waveN * 2 + nt;         // 0..7 within stage
                const char* q  = bstage + (((kc * 8 + ntgl) * 32 + lane) * 16) * 2;
                b[nt].q[0] = *(const float4*)(q);
                b[nt].q[1] = *(const float4*)(q + 16);
            }
#pragma unroll
            for (int mt = 0; mt < 4; ++mt)
#pragma unroll
                for (int nt = 0; nt < 2; ++nt)
                    acc[mt][nt] = __builtin_amdgcn_wmma_f32_16x16x32_bf16(
                        false, a[mt].v, false, b[nt].v,
                        (short)0, acc[mt][nt], false, false);
        }
        waitB();   // tap+1 staged & all waves done with this tap's buffer
    }

    // Epilogue: transpose each 16x16 f32 tile through per-wave LDS, fused
    // ReLU, M-contiguous NCHW stores. Fully unrolled (keep acc in VGPRs).
    float* wlds = (float*)(smem + 2 * kTapBytes) + wave * 256;
#pragma unroll
    for (int mt = 0; mt < 4; ++mt) {
#pragma unroll
        for (int nt = 0; nt < 2; ++nt) {
#pragma unroll
            for (int r = 0; r < 8; ++r) {
                const int m = r + hiA * 8;      // C layout: VGPR r -> M=r / r+8
                wlds[m * 16 + lrow] = acc[mt][nt][r];
            }
            const int n = nIdx[mt], hw = hwIdx[mt];
            const int co0 = coBase + nt * 16;
#pragma unroll
            for (int e = 0; e < 8; ++e) {
                const int nn = e + hiA * 8;
                float v = wlds[lrow * 16 + nn];
                v = v > 0.f ? v : 0.f;
                out[((size_t)n * kC + (co0 + nn)) * kHW + hw] = v;
            }
        }
    }
}

// ---------------- host launcher ----------------
extern "C" void kernel_launch(void* const* d_in, const int* in_sizes, int n_in,
                              void* d_out, int out_size, void* d_ws, size_t ws_size,
                              hipStream_t stream) {
    const float* x     = (const float*)d_in[0];
    const float* gamma = (const float*)d_in[1];
    const float* beta  = (const float*)d_in[2];
    const float* Wsrc  = (const float*)d_in[3];
    float* out = (float*)d_out;

    char* ws = (char*)d_ws;
    const size_t xpadElems = (size_t)kNB * kHP * kWP * kC;    // ~55.1M bf16
    const size_t wbElems   = (size_t)kKTOT * kC;              // 589824 bf16
    __bf16* xpad  = (__bf16*)ws;
    __bf16* Wb    = (__bf16*)(ws + xpadElems * sizeof(__bf16));
    float*  scale = (float*)(ws + (xpadElems + wbElems) * sizeof(__bf16));
    float*  shift = scale + kC;

    // zero padded activation buffer (borders stay zero)
    hipMemsetAsync(xpad, 0, xpadElems * sizeof(__bf16), stream);

    bn_stats_kernel<<<kC, 256, 0, stream>>>(x, gamma, beta, scale, shift);

    binarize_kernel<<<dim3(kMTOT / 256, kC / 8), 256, 0, stream>>>(
        x, scale, shift, xpad);

    wcvt_kernel<<<(72 * 16 * 32) / 256, 256, 0, stream>>>(Wsrc, Wb);

    const int dynLds = 2 * kTapBytes + 8 * 256 * (int)sizeof(float); // 139264
    wmma_conv_kernel<<<dim3(kMTOT / 128, kC / 128), 256, dynLds, stream>>>(
        xpad, Wb, out);
}
